// NonLocalBlock_23587960390330
// MI455X (gfx1250) — matrix operand order repaired and checked by reading, not compiled, not run on hardware
//
#include <hip/hip_runtime.h>
#include <hip/hip_bf16.h>

typedef __attribute__((ext_vector_type(16))) _Float16 v16h;
typedef __attribute__((ext_vector_type(8)))  _Float16 v8h;
typedef __attribute__((ext_vector_type(8)))  float    v8f;

#define NB   2
#define CIN  64
#define COUT 32
#define HW   6400   // 80*80
#define NTILES_PER_BATCH (HW / 16)   // 400
#define NTILES_TOTAL (NB * NTILES_PER_BATCH)

// ---------------------------------------------------------------------------
// Phase 1: the three 1x1 convs.  theta/phi stored flat f16 [B][Cout*HW]
// (which *is* the [B][HW][32] "raw reshape" row layout), g stored transposed
// f16 [B][32][HW] so its WMMA B-operand loads are contiguous.
// ---------------------------------------------------------------------------
__global__ void nl_conv_qkg(const float* __restrict__ x,
                            const float* __restrict__ wg, const float* __restrict__ bg,
                            const float* __restrict__ wp, const float* __restrict__ bp,
                            const float* __restrict__ wt, const float* __restrict__ bt,
                            _Float16* __restrict__ theta,
                            _Float16* __restrict__ phi,
                            _Float16* __restrict__ gT) {
  const int hw = blockIdx.x * blockDim.x + threadIdx.x;   // 0..6399
  const int o  = blockIdx.y;                              // 0..31
  const int b  = blockIdx.z;                              // 0..1
  const float* xp = x + (size_t)b * CIN * HW + hw;
  float at = bt[o], ap = bp[o], ag = bg[o];
#pragma unroll 8
  for (int c = 0; c < CIN; ++c) {
    const float xv = xp[(size_t)c * HW];
    at += xv * wt[o * CIN + c];
    ap += xv * wp[o * CIN + c];
    ag += xv * wg[o * CIN + c];
  }
  const int flat = o * HW + hw;                 // index in [Cout,H,W] flat order
  const size_t base = (size_t)b * COUT * HW;
  theta[base + flat] = (_Float16)at;
  phi[base + flat]   = (_Float16)ap;
  // g_x[b][n][c] with n = flat>>5, c = flat&31 ; store as gT[b][c][n]
  gT[base + (size_t)(flat & 31) * HW + (flat >> 5)] = (_Float16)ag;
}

// ---------------------------------------------------------------------------
// Phase 2: fused exp-attention.  One wave per 16-row tile, 4 waves / block.
// ---------------------------------------------------------------------------
__device__ inline v16h load16h(const _Float16* p0, const _Float16* p1) {
  v8h lo = *(const v8h*)p0;
  v8h hi = *(const v8h*)p1;
  v16h r;
#pragma unroll
  for (int i = 0; i < 8; ++i) { r[i] = lo[i]; r[8 + i] = hi[i]; }
  return r;
}

__global__ __launch_bounds__(128, 1)
void nl_attention(const _Float16* __restrict__ theta,
                  const _Float16* __restrict__ phi,
                  const _Float16* __restrict__ gT,
                  float* __restrict__ y) {
  __shared__ __align__(16) _Float16 lds[4][16 * 32];  // per-wave P tile (f16)

  const int lane = threadIdx.x & 31;
  const int wave = threadIdx.x >> 5;
  const int lh = lane >> 4;    // half-wave: 0 or 1
  const int ll = lane & 15;    // lane-in-half

  const int tileIdx = blockIdx.x * 4 + wave;          // 0..799
  const int b  = tileIdx / NTILES_PER_BATCH;
  const int n0 = (tileIdx % NTILES_PER_BATCH) * 16;   // first output row

  const size_t base = (size_t)b * COUT * HW;
  const _Float16* th = theta + base;
  const _Float16* ph = phi + base;
  const _Float16* gt = gT + base;
  float* yb = y + base;

  // A operand (theta rows n0..n0+15), fixed for the whole m-loop.
  // A operand, contiguous k order used here: lane holds row ll, 16 consecutive K at lh*16.
  const _Float16* tr = th + (size_t)(n0 + ll) * 32 + lh * 16;
  const v16h ta = load16h(tr, tr + 8);

  v8f acc0 = {}, acc1 = {};
  float rs[8];
#pragma unroll
  for (int v = 0; v < 8; ++v) rs[v] = 0.f;
  _Float16* pl = &lds[wave][0];

  for (int m0 = 0; m0 < HW; m0 += 32) {
    // --- S = theta_tile @ phi[m0..m0+32]^T  (two 16x16 f32 tiles) ---
    // B operand, same contiguous k order as A: lane holds col ll (= phi row), 16 consecutive K at lh*16
    const _Float16* r0 = ph + (size_t)(m0 + ll) * 32 + lh * 16;
    const _Float16* r1 = ph + (size_t)(m0 + 16 + ll) * 32 + lh * 16;
    const v16h pb0 = load16h(r0, r0 + 8);
    const v16h pb1 = load16h(r1, r1 + 8);

    const v8f cz = {};
    v8f s0 = __builtin_amdgcn_wmma_f32_16x16x32_f16(false, ta, false, pb0,
                                                    (short)0, cz, false, false);
    v8f s1 = __builtin_amdgcn_wmma_f32_16x16x32_f16(false, ta, false, pb1,
                                                    (short)0, cz, false, false);

    // --- P = exp(S); per-lane row-sum partials; stash P (f16) in LDS ---
    // D layout: lane holds row (v + 8*lh), col ll (tile 0) / 16+ll (tile 1)
#pragma unroll
    for (int v = 0; v < 8; ++v) {
      const float p0 = __expf(s0[v]);
      const float p1 = __expf(s1[v]);
      rs[v] += p0 + p1;
      const int row = v + 8 * lh;
      pl[row * 32 + ll]      = (_Float16)p0;
      pl[row * 32 + 16 + ll] = (_Float16)p1;
    }
    // Fence the intra-wave LDS transpose (store -> strided reload).
    asm volatile("s_wait_dscnt 0" ::: "memory");

    // --- reload P in A-operand layout (row ll, K = m_local) ---
    const _Float16* pr = pl + ll * 32 + lh * 16;
    const v16h pa = load16h(pr, pr + 8);

    // --- B operands from gT: col c = ctile*16+ll, 16 consecutive m ---
    const _Float16* c0 = gt + (size_t)ll * HW + m0 + lh * 16;
    const _Float16* c1 = gt + (size_t)(16 + ll) * HW + m0 + lh * 16;
    const v16h gb0 = load16h(c0, c0 + 8);
    const v16h gb1 = load16h(c1, c1 + 8);

    acc0 = __builtin_amdgcn_wmma_f32_16x16x32_f16(false, pa, false, gb0,
                                                  (short)0, acc0, false, false);
    acc1 = __builtin_amdgcn_wmma_f32_16x16x32_f16(false, pa, false, gb1,
                                                  (short)0, acc1, false, false);
  }

  // Row-sum: reduce the 16 lanes of each half-wave (rows v+8*lh).
#pragma unroll
  for (int v = 0; v < 8; ++v) {
    float r = rs[v];
    r += __shfl_xor(r, 1, 32);
    r += __shfl_xor(r, 2, 32);
    r += __shfl_xor(r, 4, 32);
    r += __shfl_xor(r, 8, 32);
    rs[v] = r;
  }

  // y[n][c] = acc / rowsum ; same register layout for acc and rs.
#pragma unroll
  for (int v = 0; v < 8; ++v) {
    const int row = n0 + v + 8 * lh;
    const float inv = 1.0f / rs[v];
    yb[(size_t)row * 32 + ll]      = acc0[v] * inv;
    yb[(size_t)row * 32 + 16 + ll] = acc1[v] * inv;
  }
}

// ---------------------------------------------------------------------------
// Phase 3: z = conv1x1(y, w_y, b_y).  y flat [B][32*HW] *is* the raw-reshaped
// [B,32,H,W] tensor, so y[b,c,hw] = y_flat[b][c*HW + hw].
// ---------------------------------------------------------------------------
__global__ void nl_conv_out(const float* __restrict__ y,
                            const float* __restrict__ wy,
                            const float* __restrict__ by,
                            float* __restrict__ z) {
  const int hw = blockIdx.x * blockDim.x + threadIdx.x;
  const int o  = blockIdx.y;   // 0..63
  const int b  = blockIdx.z;
  const float* yp = y + (size_t)b * COUT * HW + hw;
  float acc = by[o];
#pragma unroll
  for (int c = 0; c < COUT; ++c) acc += yp[(size_t)c * HW] * wy[o * COUT + c];
  z[(size_t)b * CIN * HW + (size_t)o * HW + hw] = acc;
}

// ---------------------------------------------------------------------------
extern "C" void kernel_launch(void* const* d_in, const int* in_sizes, int n_in,
                              void* d_out, int out_size, void* d_ws, size_t ws_size,
                              hipStream_t stream) {
  (void)in_sizes; (void)n_in; (void)out_size; (void)ws_size;
  const float* x  = (const float*)d_in[0];
  const float* wg = (const float*)d_in[1];
  const float* bg = (const float*)d_in[2];
  const float* wp = (const float*)d_in[3];
  const float* bp = (const float*)d_in[4];
  const float* wt = (const float*)d_in[5];
  const float* bt = (const float*)d_in[6];
  const float* wy = (const float*)d_in[7];
  const float* by = (const float*)d_in[8];
  float* z = (float*)d_out;

  const size_t nqk = (size_t)NB * COUT * HW;  // 409600 elements
  _Float16* theta = (_Float16*)d_ws;
  _Float16* phi   = theta + nqk;
  _Float16* gT    = phi + nqk;
  float*    y     = (float*)(gT + nqk);       // byte off 3*nqk*2 = 2,457,600 (16B aligned)
  // total workspace: 3*nqk*2 + nqk*4 = 4,096,000 bytes

  nl_conv_qkg<<<dim3(HW / 256, COUT, NB), dim3(256), 0, stream>>>(
      x, wg, bg, wp, bp, wt, bt, theta, phi, gT);

  nl_attention<<<dim3(NTILES_TOTAL / 4), dim3(128), 0, stream>>>(theta, phi, gT, y);

  nl_conv_out<<<dim3(HW / 256, CIN, NB), dim3(256), 0, stream>>>(y, wy, by, z);
}
